// JointNet_noGIDX_9474697855031
// MI455X (gfx1250) — matrix operand order repaired and checked by reading, not compiled, not run on hardware
//
#include <hip/hip_runtime.h>

#define B_     4
#define N_     4096
#define J_     24
#define KR_    4
#define KNN_   20
#define SLOPE_ 0.2f

typedef __attribute__((ext_vector_type(16))) __bf16    v16bf;
typedef __attribute__((ext_vector_type(8)))  float     v8f;
typedef __attribute__((ext_vector_type(4)))  unsigned  uint4v;

union Frag16 { v16bf v; uint4v q[2]; unsigned d[8]; };

// round-to-nearest-even f32 -> bf16 bit pattern
__device__ __forceinline__ unsigned short f2bf_bits(float f) {
  union { float f; unsigned u; } x; x.f = f;
  unsigned r = x.u + 0x7FFFu + ((x.u >> 16) & 1u);
  return (unsigned short)(r >> 16);
}

// ---------------------------------------------------------------- utilities
// f32 [rows,C] -> bf16 [rows,Cp], zero-padded K
__global__ void cvtpad_kernel(const float* __restrict__ src,
                              unsigned short* __restrict__ dst,
                              int rows, int C, int Cp) {
  long t = (long)blockIdx.x * blockDim.x + threadIdx.x;
  if (t >= (long)rows * Cp) return;
  int c = (int)(t % Cp);
  long r = t / Cp;
  dst[t] = (c < C) ? f2bf_bits(src[r * C + c]) : (unsigned short)0;
}

// w [O,2C] -> bf16 [O, 2*Cp].
// region0 (k<Cp):  w_d[o,k]                (weights on (nbr - ctr))
// region1 (k>=Cp): w_c[o,k] - w_d[o,k]     (so raw [nbr | ctr] rows can be used)
__global__ void wpad_kernel(const float* __restrict__ w,
                            unsigned short* __restrict__ wp,
                            int O, int C, int Cp) {
  long t = (long)blockIdx.x * blockDim.x + threadIdx.x;
  if (t >= (long)O * 2 * Cp) return;
  int kp = (int)(t % (2 * Cp));
  long o = t / (2 * Cp);
  int reg = (kp < Cp) ? 0 : 1;
  int kc  = kp - reg * Cp;
  unsigned short v = 0;
  if (kc < C) {
    float wd = w[o * (2 * C) + kc];
    float val = (reg == 0) ? wd : (w[o * (2 * C) + C + kc] - wd);
    v = f2bf_bits(val);
  }
  wp[t] = v;
}

__global__ void norms_kernel(const float* __restrict__ x, float* __restrict__ xx,
                             int npts, int C) {
  int t = blockIdx.x * blockDim.x + threadIdx.x;
  if (t >= npts) return;
  const float* row = x + (size_t)t * C;
  float s = 0.f;
  for (int c = 0; c < C; c++) { float v = row[c]; s += v * v; }
  xx[t] = s;
}

// ------------------------------------------------- fused dist-GEMM + top-k
// One wave per 16-row strip; A fragments live in registers across all column
// tiles; inner loop is 2 b128 loads + 1 WMMA per K-chunk.
template <int CP>
__global__ void __launch_bounds__(32)
knn_kernel(const unsigned short* __restrict__ xbf, const float* __restrict__ xx,
           int* __restrict__ idx, int N) {
  constexpr int CH = CP / 32;
  __shared__ float tile[256];
  const int l = threadIdx.x;
  const int half = l >> 4;
  const int lane15 = l & 15;
  const int b  = blockIdx.y;
  const int ib = blockIdx.x;
  const size_t bN = (size_t)b * N;
  const float* xxb = xx + bN;
  const int irow = ib * 16 + lane15;
  const uint4v* arow = (const uint4v*)(xbf + (bN + irow) * CP);

  // hoist this strip's A fragments into registers (invariant over jt)
  Frag16 af[CH];
#pragma unroll
  for (int ch = 0; ch < CH; ch++) {
    af[ch].q[0] = arow[ch * 4 + half * 2];       // K-run 0..7   (this half)
    af[ch].q[1] = arow[ch * 4 + half * 2 + 1];   // K-run 8..15  (this half)
  }
  float xi[8];
#pragma unroll
  for (int v = 0; v < 8; v++) xi[v] = xxb[ib * 16 + v + (half << 3)];

  float topd[KNN_]; int topi[KNN_];
#pragma unroll
  for (int t = 0; t < KNN_; t++) { topd[t] = 3.4e38f; topi[t] = 0; }

  for (int jt = 0; jt < (N >> 4); jt++) {
    const uint4v* brow = (const uint4v*)(xbf + (bN + jt * 16 + lane15) * CP);
    v8f c = {};
#pragma unroll
    for (int ch = 0; ch < CH; ch++) {
      Frag16 bb;
      bb.q[0] = brow[ch * 4 + half * 2];     // B: K-run 0..15 (this half)
      bb.q[1] = brow[ch * 4 + half * 2 + 1];
      c = __builtin_amdgcn_wmma_f32_16x16x32_bf16(false, af[ch].v, false, bb.v,
                                                  (short)0, c, false, false);
    }
    float xxj = xxb[jt * 16 + lane15];
#pragma unroll
    for (int v = 0; v < 8; v++) {
      int m = v + (half << 3);
      tile[m * 16 + lane15] = xi[v] + xxj - 2.0f * c[v];
    }
    asm volatile("s_wait_dscnt 0" ::: "memory");
    if (l < 16) {
      for (int t = 0; t < 16; t++) {          // runtime loop: slim static code
        float d = tile[l * 16 + t];
        int j = jt * 16 + t;
        if (d < topd[KNN_ - 1]) {
#pragma unroll
          for (int q = KNN_ - 1; q > 0; q--) {
            bool sh = topd[q - 1] > d;
            topd[q] = sh ? topd[q - 1] : topd[q];
            topi[q] = sh ? topi[q - 1] : topi[q];
          }
#pragma unroll
          for (int q = KNN_ - 1; q >= 0; q--) {
            bool up = (q == 0) ? true : (topd[q - 1] <= d);
            if (topd[q] > d && up) { topd[q] = d; topi[q] = j; }
          }
        }
      }
    }
    asm volatile("s_wait_dscnt 0" ::: "memory");
  }
  if (l < 16) {
#pragma unroll
    for (int t = 0; t < KNN_; t++)
      idx[(bN + ib * 16 + l) * KNN_ + t] = topi[t];
  }
}

// ----------------------------------------------------- WMMA edge convolution
// out[b,p,o] = max_kk leaky( nbr.w_d + ctr.(w_c - w_d) + bias )
// The ctr term is neighbor-invariant: computed once, used as accumulator seed.
__global__ void __launch_bounds__(32)
edge_conv_kernel(const unsigned short* __restrict__ xbf,
                 const int* __restrict__ idx,
                 const unsigned short* __restrict__ wpad,
                 const float* __restrict__ bias, float* __restrict__ out,
                 int N, int Cp, int O, int k) {
  const int l = threadIdx.x;
  const int half = l >> 4;
  const int lane15 = l & 15;
  const int b  = blockIdx.z;
  const int mt = blockIdx.x;
  const int ot = blockIdx.y;
  const int chunks = Cp >> 5;
  const size_t bN = (size_t)b * N;

  const int p = mt * 16 + lane15;
  const bool valid = p < N;
  const int pc = valid ? p : (N - 1);       // clamp: loads always legal
  const uint4v* ctr = (const uint4v*)(xbf + (bN + pc) * Cp);
  const uint4v* wrow = (const uint4v*)(wpad + (size_t)(ot * 16 + lane15) * (2 * Cp));
  const float bo = bias[ot * 16 + lane15];

  // one-time neighbor-invariant part: ctr . (w_c - w_d)
  v8f cc = {};
  for (int ch = 0; ch < chunks; ch++) {
    Frag16 a, bb;
    a.q[0]  = ctr[ch * 4 + half * 2];
    a.q[1]  = ctr[ch * 4 + half * 2 + 1];
    bb.q[0] = wrow[(chunks + ch) * 4 + half * 2];
    bb.q[1] = wrow[(chunks + ch) * 4 + half * 2 + 1];
    cc = __builtin_amdgcn_wmma_f32_16x16x32_bf16(false, a.v, false, bb.v,
                                                 (short)0, cc, false, false);
  }

  v8f mx;
#pragma unroll
  for (int v = 0; v < 8; v++) mx[v] = -3.4e38f;

  for (int kk = 0; kk < k; kk++) {
    int nbr = idx[(bN + pc) * k + kk];
    const uint4v* nb = (const uint4v*)(xbf + (bN + nbr) * Cp);
    v8f c = cc;                              // seed with the invariant term
    for (int ch = 0; ch < chunks; ch++) {
      Frag16 a, bb;
      a.q[0]  = nb[ch * 4 + half * 2];
      a.q[1]  = nb[ch * 4 + half * 2 + 1];
      bb.q[0] = wrow[ch * 4 + half * 2];
      bb.q[1] = wrow[ch * 4 + half * 2 + 1];
      c = __builtin_amdgcn_wmma_f32_16x16x32_bf16(false, a.v, false, bb.v,
                                                  (short)0, c, false, false);
    }
#pragma unroll
    for (int v = 0; v < 8; v++) {
      float h = c[v] + bo;
      h = (h >= 0.f) ? h : SLOPE_ * h;
      mx[v] = fmaxf(mx[v], h);
    }
  }
#pragma unroll
  for (int v = 0; v < 8; v++) {
    int m = v + (half << 3);
    int pp = mt * 16 + m;
    if (pp < N) out[(bN + pp) * O + ot * 16 + lane15] = mx[v];
  }
}

// ------------------------------------------------------------ small kernels
__global__ void pack_feat_kernel(const float* __restrict__ V,
                                 const float* __restrict__ h1,
                                 const float* __restrict__ h2,
                                 const float* __restrict__ h3,
                                 float* __restrict__ feat) {
  long t = (long)blockIdx.x * blockDim.x + threadIdx.x;
  const long total = (long)B_ * N_ * 451;
  if (t >= total) return;
  int c = (int)(t % 451);
  long pn = t / 451;
  float v;
  if (c < 3)        v = V [pn * 3   + c];
  else if (c < 67)  v = h1[pn * 64  + (c - 3)];
  else if (c < 195) v = h2[pn * 128 + (c - 67)];
  else              v = h3[pn * 256 + (c - 195)];
  feat[t] = v;
}

__global__ void pool_kernel(const float* __restrict__ W,
                            const float* __restrict__ feat,
                            float* __restrict__ pooled) {
  __shared__ float red[256];
  const int j = blockIdx.x, b = blockIdx.y, tid = threadIdx.x;
  const float* wrow = W + ((size_t)b * J_ + j) * N_;
  float s = 0.f;
  for (int n = tid; n < N_; n += 256) s += wrow[n];
  red[tid] = s; __syncthreads();
  for (int st = 128; st > 0; st >>= 1) {
    if (tid < st) red[tid] += red[tid + st];
    __syncthreads();
  }
  float inv = 1.0f / (red[0] + 1e-5f);
  for (int c = tid; c < 451; c += 256) {
    float acc = 0.f;
    const float* fb = feat + (size_t)b * N_ * 451 + c;
    for (int n = 0; n < N_; n++) acc += wrow[n] * fb[(size_t)n * 451];
    pooled[((size_t)b * J_ + j) * 451 + c] = acc * inv;
  }
}

__global__ void pack_joints_kernel(const float* __restrict__ v1,
                                   const float* __restrict__ v2,
                                   const float* __restrict__ v3,
                                   float* __restrict__ joints) {
  int t = blockIdx.x * blockDim.x + threadIdx.x;
  const int total = B_ * J_ * 448;
  if (t >= total) return;
  int c = t % 448; int r = t / 448;
  float v;
  if (c < 256)      v = v1[r * 256 + c];
  else if (c < 384) v = v2[r * 128 + (c - 256)];
  else              v = v3[r * 64  + (c - 384)];
  joints[t] = v;
}

__global__ void mlp_kernel(const float* __restrict__ in, const float* __restrict__ w,
                           const float* __restrict__ bias, float* __restrict__ out,
                           int rows, int Cin, int O, int leaky) {
  int t = blockIdx.x * blockDim.x + threadIdx.x;
  if (t >= rows * O) return;
  int o = t % O, r = t / O;
  float acc = bias[o];
  const float* ir = in + (size_t)r * Cin;
  const float* wr = w + (size_t)o * Cin;
  for (int c = 0; c < Cin; c++) acc += ir[c] * wr[c];
  if (leaky) acc = (acc >= 0.f) ? acc : SLOPE_ * acc;
  out[t] = acc;
}

// ----------------------------------------------------------------- launcher
extern "C" void kernel_launch(void* const* d_in, const int* in_sizes, int n_in,
                              void* d_out, int out_size, void* d_ws, size_t ws_size,
                              hipStream_t stream) {
  (void)in_sizes; (void)n_in; (void)out_size; (void)ws_size;
  const float* V    = (const float*)d_in[0];
  const float* W    = (const float*)d_in[1];
  const int*   sIdx = (const int*)d_in[2];
  const float* g1w = (const float*)d_in[3],  *g1b = (const float*)d_in[4];
  const float* g2w = (const float*)d_in[5],  *g2b = (const float*)d_in[6];
  const float* g3w = (const float*)d_in[7],  *g3b = (const float*)d_in[8];
  const float* s1w = (const float*)d_in[9],  *s1b = (const float*)d_in[10];
  const float* s2w = (const float*)d_in[11], *s2b = (const float*)d_in[12];
  const float* s3w = (const float*)d_in[13], *s3b = (const float*)d_in[14];
  const float* m1w = (const float*)d_in[15], *m1b = (const float*)d_in[16];
  const float* m2w = (const float*)d_in[17], *m2b = (const float*)d_in[18];
  const float* m3w = (const float*)d_in[19], *m3b = (const float*)d_in[20];

  char* ws = (char*)d_ws;
  size_t off = 0;
  auto take = [&](size_t bytes) -> char* {
    char* p = ws + off;
    off += (bytes + 255) & ~(size_t)255;
    return p;
  };
  const int npts = B_ * N_;
  const int nj   = B_ * J_;
  unsigned short* xb   = (unsigned short*)take((size_t)npts * 128 * 2); // per-layer reuse
  unsigned short* sb   = (unsigned short*)take((size_t)nj * 480 * 2);   // skeleton reuse
  float* xx     = (float*)take((size_t)npts * 4);
  int*   idx    = (int*)  take((size_t)npts * KNN_ * 4);
  float* h1     = (float*)take((size_t)npts * 64  * 4);
  float* h2     = (float*)take((size_t)npts * 128 * 4);
  float* h3     = (float*)take((size_t)npts * 256 * 4);
  float* feat   = (float*)take((size_t)npts * 451 * 4);
  float* pooled = (float*)take((size_t)nj * 451 * 4);
  float* v1     = (float*)take((size_t)nj * 256 * 4);
  float* v2     = (float*)take((size_t)nj * 128 * 4);
  float* v3     = (float*)take((size_t)nj * 64  * 4);
  float* joints = (float*)take((size_t)nj * 448 * 4);
  float* t1     = (float*)take((size_t)nj * 512 * 4);
  float* t2     = (float*)take((size_t)nj * 256 * 4);
  unsigned short* wg1 = (unsigned short*)take((size_t)64  * 64  * 2);
  unsigned short* wg2 = (unsigned short*)take((size_t)128 * 128 * 2);
  unsigned short* wg3 = (unsigned short*)take((size_t)256 * 256 * 2);
  unsigned short* wk1 = (unsigned short*)take((size_t)256 * 960 * 2);
  unsigned short* wk2 = (unsigned short*)take((size_t)128 * 512 * 2);
  unsigned short* wk3 = (unsigned short*)take((size_t)64  * 256 * 2);

  auto wpad = [&](const float* s, unsigned short* d, int O, int C, int Cp) {
    long n = (long)O * 2 * Cp;
    wpad_kernel<<<(int)((n + 255) / 256), 256, 0, stream>>>(s, d, O, C, Cp);
  };
  auto cvtp = [&](const float* s, unsigned short* d, int rows, int C, int Cp) {
    long n = (long)rows * Cp;
    cvtpad_kernel<<<(int)((n + 255) / 256), 256, 0, stream>>>(s, d, rows, C, Cp);
  };
  wpad(g1w, wg1, 64, 3,   32);  wpad(g2w, wg2, 128, 64,  64);
  wpad(g3w, wg3, 256, 128, 128);
  wpad(s1w, wk1, 256, 451, 480); wpad(s2w, wk2, 128, 256, 256);
  wpad(s3w, wk3, 64, 128, 128);

  // ---- geoNet layer 1 (C=3, Cp=32 -> O=64)
  cvtp(V, xb, npts, 3, 32);
  norms_kernel<<<(npts + 255) / 256, 256, 0, stream>>>(V, xx, npts, 3);
  knn_kernel<32><<<dim3(N_ / 16, B_), 32, 0, stream>>>(xb, xx, idx, N_);
  edge_conv_kernel<<<dim3(N_ / 16, 4, B_), 32, 0, stream>>>(xb, idx, wg1, g1b, h1, N_, 32, 64, KNN_);
  // ---- geoNet layer 2 (C=64 -> O=128)
  cvtp(h1, xb, npts, 64, 64);
  norms_kernel<<<(npts + 255) / 256, 256, 0, stream>>>(h1, xx, npts, 64);
  knn_kernel<64><<<dim3(N_ / 16, B_), 32, 0, stream>>>(xb, xx, idx, N_);
  edge_conv_kernel<<<dim3(N_ / 16, 8, B_), 32, 0, stream>>>(xb, idx, wg2, g2b, h2, N_, 64, 128, KNN_);
  // ---- geoNet layer 3 (C=128 -> O=256)
  cvtp(h2, xb, npts, 128, 128);
  norms_kernel<<<(npts + 255) / 256, 256, 0, stream>>>(h2, xx, npts, 128);
  knn_kernel<128><<<dim3(N_ / 16, B_), 32, 0, stream>>>(xb, xx, idx, N_);
  edge_conv_kernel<<<dim3(N_ / 16, 16, B_), 32, 0, stream>>>(xb, idx, wg3, g3b, h3, N_, 128, 256, KNN_);

  // ---- concat + weighted pooling
  long featTotal = (long)npts * 451;
  pack_feat_kernel<<<(int)((featTotal + 255) / 256), 256, 0, stream>>>(V, h1, h2, h3, feat);
  pool_kernel<<<dim3(J_, B_), 256, 0, stream>>>(W, feat, pooled);

  // ---- skeleton edge convs (fixed one-ring graph, k=4)
  cvtp(pooled, sb, nj, 451, 480);
  edge_conv_kernel<<<dim3(2, 16, B_), 32, 0, stream>>>(sb, sIdx, wk1, s1b, v1, J_, 480, 256, KR_);
  cvtp(v1, sb, nj, 256, 256);
  edge_conv_kernel<<<dim3(2, 8,  B_), 32, 0, stream>>>(sb, sIdx, wk2, s2b, v2, J_, 256, 128, KR_);
  cvtp(v2, sb, nj, 128, 128);
  edge_conv_kernel<<<dim3(2, 4,  B_), 32, 0, stream>>>(sb, sIdx, wk3, s3b, v3, J_, 128, 64, KR_);

  // ---- joint MLP
  pack_joints_kernel<<<(nj * 448 + 255) / 256, 256, 0, stream>>>(v1, v2, v3, joints);
  mlp_kernel<<<(nj * 512 + 255) / 256, 256, 0, stream>>>(joints, m1w, m1b, t1, nj, 448, 512, 1);
  mlp_kernel<<<(nj * 256 + 255) / 256, 256, 0, stream>>>(t1, m2w, m2b, t2, nj, 512, 256, 1);
  mlp_kernel<<<(nj * 3 + 255) / 256, 256, 0, stream>>>(t2, m3w, m3b, (float*)d_out, nj, 256, 3, 0);
}
